// CustomGRU_30700426232290
// MI455X (gfx1250) — matrix operand order repaired and checked by reading, not compiled, not run on hardware
//
#include <hip/hip_runtime.h>
#include <math.h>

// ---------------------------------------------------------------------------
// Persistent-WMMA GRU for MI455X (gfx1250, wave32).
//
//   B=64, S=512, F=512, H=1024.
//   64 workgroups; WG `wg` owns H-columns [16*wg, 16*wg+16), i.e. gate
//   columns {j, H+j, 2H+j}. Its bf16 weight slab (W_ih 512x48 + W_hh 1024x48
//   = 144 KB) lives in LDS for the whole kernel, pre-swizzled into WMMA
//   B-fragment order. 128 threads = 4 wave32s; wave w computes batch rows
//   [16w,16w+16) with v_wmma_f32_16x16x32_bf16 (f32 accumulate).
//   h is double-buffered in workspace as bf16; one atomic grid barrier per
//   timestep. Assumes >=64 WGPs co-resident (MI455X has far more).
// ---------------------------------------------------------------------------

#define B_ 64
#define S_ 512
#define F_ 512
#define H_ 1024
#define NWG 64           // H_/16
#define NTHREADS 128     // 4 waves
#define SLAB_ELEMS (144 * 512)       // 48 x-frags + 96 h-frags, 512 bf16 each
#define LDS_BYTES  (SLAB_ELEMS * 2)  // 147456 = 144 KB

typedef __attribute__((ext_vector_type(8)))  __bf16 v8bf;
typedef __attribute__((ext_vector_type(16))) __bf16 v16bf;
typedef __attribute__((ext_vector_type(8)))  float  v8f;

static __device__ __forceinline__ __bf16 f2bf(float f) {
  unsigned u = __builtin_bit_cast(unsigned, f);
  unsigned r = u + 0x7FFFu + ((u >> 16) & 1u);   // round-to-nearest-even
  unsigned short h = (unsigned short)(r >> 16);
  return __builtin_bit_cast(__bf16, h);
}
static __device__ __forceinline__ float bf2f(__bf16 b) {
  unsigned short h = __builtin_bit_cast(unsigned short, b);
  unsigned u = ((unsigned)h) << 16;
  return __builtin_bit_cast(float, u);
}
static __device__ __forceinline__ float sigmoidf_(float x) {
  return 1.0f / (1.0f + __expf(-x));
}
static __device__ __forceinline__ v16bf cat8(v8bf a, v8bf b) {
  return __builtin_shufflevector(a, b, 0,1,2,3,4,5,6,7,8,9,10,11,12,13,14,15);
}

// A-fragment (16x32 bf16) from a row-major bf16 row pointer.
// Lanes 0-15: M=lane, K = {0..7, 16..23}; lanes 16-31: M=lane-16, K={8..15,24..31}.
static __device__ __forceinline__ v16bf loadA(const __bf16* rowp, int segA) {
  v8bf a0 = *(const v8bf*)(rowp + segA);        // 16B global load
  v8bf a1 = *(const v8bf*)(rowp + segA + 16);   // 16B global load
  return cat8(a0, a1);
}
// B-fragment (32x16 bf16) from LDS slab in pre-swizzled order:
// lane l holds column N=l%16, K = (l<16 ? {0..7,16..23} : {8..15,24..31}), same as loadA.
static __device__ __forceinline__ v16bf loadB(const __bf16* p) {
  v8bf b0 = *(const v8bf*)(p);       // ds_load_b128
  v8bf b1 = *(const v8bf*)(p + 8);   // ds_load_b128
  return cat8(b0, b1);
}

#define WMMA_BF16(A, Bm, C) \
  __builtin_amdgcn_wmma_f32_16x16x32_bf16(false, (A), false, (Bm), (short)0, (C), false, false)

// --- pre-pass: zero barrier counter + h0 buffer ----------------------------
__global__ void gru_init_ws(unsigned* __restrict__ p, int n) {
  int i = blockIdx.x * blockDim.x + threadIdx.x;
  int stride = gridDim.x * blockDim.x;
  for (; i < n; i += stride) p[i] = 0u;
}

// --- pre-pass: convert x to bf16 once (cuts per-step VALU + bytes) ---------
__global__ void gru_xcvt(const float* __restrict__ x, __bf16* __restrict__ xb, int n) {
  int i = blockIdx.x * blockDim.x + threadIdx.x;
  int stride = gridDim.x * blockDim.x;
  for (; i < n; i += stride) xb[i] = f2bf(x[i]);
}

// --- main persistent kernel ------------------------------------------------
__global__ __launch_bounds__(NTHREADS, 1)
void gru_persistent(const __bf16* __restrict__ xb,     // [B,S,F] bf16
                    const float*  __restrict__ wih,    // [F,3H] f32
                    const float*  __restrict__ whh,    // [H,3H] f32
                    const float*  __restrict__ bias,   // [3H]
                    float*        __restrict__ out,    // [B,S,H] ++ [B,H]
                    __bf16*       __restrict__ hb0,    // [B,H] bf16 (zeroed)
                    __bf16*       __restrict__ hb1,    // [B,H] bf16
                    unsigned*     __restrict__ bar) {
  extern __shared__ __align__(16) unsigned char smem_raw[];
  __bf16* slab = (__bf16*)smem_raw;

  const int tid = threadIdx.x;
  const int wg  = blockIdx.x;          // owns H cols [wg*16, wg*16+16)

  // ---- prologue: fill LDS slab in B-fragment order (f32 -> bf16) ----------
  // frag < 48  : x-side,  frag = g*16 + kt  (g=gate, kt over F/32)
  // frag >= 48 : h-side,  frag-48 = g*32 + kt (kt over H/32)
  for (int e = tid; e < SLAB_ELEMS; e += NTHREADS) {
    int frag = e >> 9;
    int rem  = e & 511;
    int l    = rem >> 4;                 // dest lane slot
    int i    = rem & 15;
    int kl   = ((l < 16) ? 0 : 8) + (i & 7) + ((i >> 3) << 4);  // K within 32-tile
    int ncol = l & 15;                   // N within 16-tile
    float w;
    if (frag < 48) {
      int g = frag >> 4, kt = frag & 15;
      w = wih[(size_t)(kt * 32 + kl) * (3 * H_) + g * H_ + wg * 16 + ncol];
    } else {
      int f2 = frag - 48;
      int g = f2 >> 5, kt = f2 & 31;
      w = whh[(size_t)(kt * 32 + kl) * (3 * H_) + g * H_ + wg * 16 + ncol];
    }
    slab[e] = f2bf(w);
  }
  __syncthreads();

  const int lane   = tid & 31;
  const int wv     = tid >> 5;              // wave id == M-tile (batch rows)
  const int b0     = wv * 16;
  const int lr     = lane & 15;             // A-fragment row within tile
  const int segA   = (lane < 16) ? 0 : 8;   // A-fragment K segment start
  const int rowAdd = (lane < 16) ? 0 : 8;   // C/D row offset for this lane
  const int nc     = lane & 15;             // C/D column within tile
  const int hc     = wg * 16 + nc;          // owned H column for outputs

  const float bz = bias[hc];
  const float br = bias[H_ + hc];
  const float be = bias[2 * H_ + hc];

  const __bf16* xs = slab;                  // x-side slab (48 frags)
  const __bf16* hs = slab + 48 * 512;       // h-side slab (96 frags)

  for (int t = 0; t < S_; ++t) {
    const __bf16* hprev = (t & 1) ? hb1 : hb0;
    __bf16*       hnext = (t & 1) ? hb0 : hb1;

    v8f az = {}, ar = {}, aex = {}, aeh = {};

    // ---- x-side GEMM: [16,512] x [512,48] --------------------------------
    const __bf16* xrow = xb + ((size_t)(b0 + lr) * S_ + t) * F_;
#pragma unroll
    for (int kt = 0; kt < 16; ++kt) {
      v16bf a  = loadA(xrow + kt * 32, segA);
      v16bf wz = loadB(xs + (size_t)(0 * 16 + kt) * 512 + lane * 16);
      v16bf wr = loadB(xs + (size_t)(1 * 16 + kt) * 512 + lane * 16);
      v16bf we = loadB(xs + (size_t)(2 * 16 + kt) * 512 + lane * 16);
      az  = WMMA_BF16(a, wz, az);
      ar  = WMMA_BF16(a, wr, ar);
      aex = WMMA_BF16(a, we, aex);
    }
    // ---- h-side GEMM: [16,1024] x [1024,48] ------------------------------
    const __bf16* hrow = hprev + (size_t)(b0 + lr) * H_;
#pragma unroll
    for (int kt = 0; kt < 32; ++kt) {
      v16bf a  = loadA(hrow + kt * 32, segA);
      v16bf wz = loadB(hs + (size_t)(0 * 32 + kt) * 512 + lane * 16);
      v16bf wr = loadB(hs + (size_t)(1 * 32 + kt) * 512 + lane * 16);
      v16bf we = loadB(hs + (size_t)(2 * 32 + kt) * 512 + lane * 16);
      az  = WMMA_BF16(a, wz, az);
      ar  = WMMA_BF16(a, wr, ar);
      aeh = WMMA_BF16(a, we, aeh);
    }

    // ---- gates + state update (C/D layout: VGPR v -> M=v(+8), N=lane%16) -
#pragma unroll
    for (int v = 0; v < 8; ++v) {
      int b    = b0 + v + rowAdd;
      float z  = sigmoidf_(az[v] + bz);
      float r  = sigmoidf_(ar[v] + br);
      float e  = tanhf(aex[v] + be + r * tanhf(aeh[v]));
      float hp = bf2f(hprev[(size_t)b * H_ + hc]);
      float hn = z * hp + (1.0f - z) * e;
      out[((size_t)b * S_ + t) * H_ + hc] = hn;       // hidden_seq
      hnext[(size_t)b * H_ + hc] = f2bf(hn);
      if (t == S_ - 1)
        out[(size_t)B_ * S_ * H_ + (size_t)b * H_ + hc] = hn;  // h_last
    }

    // ---- grid-wide generation barrier ------------------------------------
    __threadfence();
    __syncthreads();
    if (tid == 0) {
      atomicAdd(bar, 1u);
      unsigned target = (unsigned)(t + 1) * NWG;
      while (__hip_atomic_load(bar, __ATOMIC_RELAXED, __HIP_MEMORY_SCOPE_AGENT) < target) {
        __builtin_amdgcn_s_sleep(2);
      }
      __threadfence();
    }
    __syncthreads();
  }
}

// ---------------------------------------------------------------------------
extern "C" void kernel_launch(void* const* d_in, const int* in_sizes, int n_in,
                              void* d_out, int out_size, void* d_ws, size_t ws_size,
                              hipStream_t stream) {
  (void)in_sizes; (void)n_in; (void)out_size; (void)ws_size;

  const float* x    = (const float*)d_in[0];   // [B,S,F]
  const float* wih  = (const float*)d_in[1];   // [F,3H]
  const float* whh  = (const float*)d_in[2];   // [H,3H]
  const float* bias = (const float*)d_in[3];   // [3H]
  float* out = (float*)d_out;

  // workspace layout: [0..256) barrier counter | hb0 | hb1 | x_bf16
  unsigned char* ws = (unsigned char*)d_ws;
  unsigned* bar = (unsigned*)ws;
  __bf16* hb0 = (__bf16*)(ws + 256);
  __bf16* hb1 = (__bf16*)(ws + 256 + (size_t)B_ * H_ * 2);
  __bf16* xbf = (__bf16*)(ws + 256 + (size_t)2 * B_ * H_ * 2);
  // total ws use: 256 + 256KB + 32MB  (~33.8 MB)

  // reset counter + zero h0 (ws is poisoned, and persists across replays)
  gru_init_ws<<<64, 256, 0, stream>>>(bar, (256 + B_ * H_ * 2) / 4);
  // one-time f32 -> bf16 conversion of x
  gru_xcvt<<<4096, 256, 0, stream>>>(x, xbf, B_ * S_ * F_);
  // persistent scan: 64 WGs, 4 waves each, 144 KB dynamic LDS
  gru_persistent<<<NWG, NTHREADS, LDS_BYTES, stream>>>(xbf, wih, whh, bias,
                                                       out, hb0, hb1, bar);
}
